// GRU_68719477028
// MI455X (gfx1250) — matrix operand order repaired and checked by reading, not compiled, not run on hardware
//
#include <hip/hip_runtime.h>
#include <hip/hip_bf16.h>

// ---------------- problem constants ----------------
#define SEQ   2048
#define BATCH 64
#define INF   512
#define HID   512
#define KA    1024            // IN+H  (phase A K)
#define NA    1536            // 2H + H (phase A N: z|r|candx)
#define NWG   12              // persistent workgroups
#define TPB   256             // 8 waves
#define STRIDE_A 1032         // LDS row stride (KA + 8 pad), elements

typedef __attribute__((ext_vector_type(16))) __bf16 v16bf;
typedef __attribute__((ext_vector_type(8)))  __bf16 v8bf;
typedef __attribute__((ext_vector_type(8)))  float  v8f;

// ---------------- workspace layout (bytes) ----------------
#define WS_WT1   0                          // bf16 [1536][1024]  B-layout phase A
#define WS_WT2   (WS_WT1 + 1536*1024*2)     // bf16 [512][512]    B-layout phase B
#define WS_H     (WS_WT2 + 512*512*2)       // f32  [64][512]     hidden state
#define WS_Z     (WS_H   + 64*512*4)        // f32  [64][512]     z gate
#define WS_C     (WS_Z   + 64*512*4)        // f32  [64][512]     x-side cand pre-act (+bc)
#define WS_RH    (WS_C   + 64*512*4)        // bf16 [64][512]     r*h (phase B A-matrix)
#define WS_CNT   (WS_RH  + 64*512*2)        // u32 barrier counter

// ---------------- prep: bf16 weight repack + h0 + barrier reset ----------------
__global__ void gru_prep(const float* __restrict__ Wg, const float* __restrict__ Wc,
                         const float* __restrict__ h0,
                         __bf16* __restrict__ Wt1, __bf16* __restrict__ Wt2,
                         float* __restrict__ hbuf, unsigned* __restrict__ cnt) {
  int i = blockIdx.x * blockDim.x + threadIdx.x;
  const int NW1 = NA * KA;        // 1,572,864
  const int NW2 = HID * HID;      // 262,144
  if (i < NW1) {
    int n = i >> 10, k = i & 1023;      // Wt1[n][k] = W_comb[k][n]
    float v;
    if (n < 1024)      v = Wg[k * 1024 + n];                 // Wg_x | Wg_h
    else if (k < 512)  v = Wc[k * 512 + (n - 1024)];         // Wc_x
    else               v = 0.0f;                             // pad (h rows of cand-x)
    Wt1[i] = (__bf16)v;
  } else if (i < NW1 + NW2) {
    int j = i - NW1; int n = j >> 9, k = j & 511;            // Wt2[n][k] = Wc_h[k][n]
    Wt2[j] = (__bf16)Wc[(512 + k) * 512 + n];
  } else if (i < NW1 + NW2 + BATCH * HID) {
    int j = i - NW1 - NW2;
    hbuf[j] = h0[j];
  } else if (i == NW1 + NW2 + BATCH * HID) {
    *cnt = 0u;
  }
}

// ---------------- lightweight multi-WG barrier ----------------
__device__ __forceinline__ void grid_barrier(unsigned* cnt, unsigned target) {
  __syncthreads();
  if (threadIdx.x == 0) {
    __threadfence();
    atomicAdd(cnt, 1u);
    while (__hip_atomic_load(cnt, __ATOMIC_RELAXED, __HIP_MEMORY_SCOPE_AGENT) < target) {
      __builtin_amdgcn_s_sleep(2);
    }
    __threadfence();
  }
  __syncthreads();
}

__device__ __forceinline__ float fast_sigmoid(float x) {
  return 1.0f / (1.0f + __expf(-x));
}
__device__ __forceinline__ float fast_tanh(float x) {
  return 2.0f / (1.0f + __expf(-2.0f * x)) - 1.0f;
}

// join two 16-byte halves into one 16-element bf16 fragment
__device__ __forceinline__ v16bf join16(v8bf lo, v8bf hi) {
  return __builtin_shufflevector(lo, hi, 0, 1, 2, 3, 4, 5, 6, 7,
                                         8, 9, 10, 11, 12, 13, 14, 15);
}

// ---------------- persistent fused GRU scan ----------------
__global__ void __launch_bounds__(TPB, 1)
gru_scan(const float* __restrict__ x, const float* __restrict__ bg,
         const float* __restrict__ bc,
         const __bf16* __restrict__ Wt1, const __bf16* __restrict__ Wt2,
         float* __restrict__ hbuf, float* __restrict__ zbuf,
         float* __restrict__ cbuf, __bf16* __restrict__ rhbuf,
         unsigned* __restrict__ cnt, float* __restrict__ out) {
  extern __shared__ __bf16 Alds[];               // [64][STRIDE_A] bf16 = [x_t | h]
  const int tid   = threadIdx.x;
  const int lane  = tid & 31;
  const int wave  = tid >> 5;
  const int strip = blockIdx.x * 8 + wave;       // 0..95 : 16-col N-strip
  const int n0    = strip * 16;
  const int ln    = lane & 15;                   // N within strip / row group
  const int lh    = lane >> 4;                   // K-half selector
  const v8f vzero = {0.f, 0.f, 0.f, 0.f, 0.f, 0.f, 0.f, 0.f};
  unsigned target = 0;

  for (int t = 0; t < SEQ; ++t) {
    // ---- stage A = [x_t | h_prev] into LDS as bf16 (all 256 threads) ----
    const float* xt = x + (size_t)t * (BATCH * INF);
    #pragma unroll 4
    for (int c = 0; c < 64; ++c) {
      int fi  = tid + c * TPB;                   // float4 index, 16384 total
      int e   = fi << 2;
      int row = e >> 10, col = e & 1023;
      const float* src = (col < 512) ? (xt + row * 512 + col)
                                     : (hbuf + row * 512 + (col - 512));
      float4 v = *(const float4*)src;
      __bf16* dst = Alds + row * STRIDE_A + col;
      dst[0] = (__bf16)v.x; dst[1] = (__bf16)v.y;
      dst[2] = (__bf16)v.z; dst[3] = (__bf16)v.w;
    }
    __syncthreads();

    // ---- phase A: [64 x 1024] @ Wcomb[1024 x 1536], this wave owns cols n0..n0+15 ----
    v8f acc[4];
    #pragma unroll
    for (int m = 0; m < 4; ++m) acc[m] = vzero;

    const __bf16* Bbase = Wt1 + (size_t)(n0 + ln) * KA + lh * 16;
    #pragma unroll 4
    for (int kt = 0; kt < 32; ++kt) {            // K = 1024, 32 per WMMA
      v16bf bfrag = *(const v16bf*)(Bbase + kt * 32);    // 32B contiguous (B layout)
      if (kt < 30) __builtin_prefetch(Bbase + (kt + 2) * 32, 0, 3);
      const int k0 = kt * 32 + lh * 16;
      #pragma unroll
      for (int m = 0; m < 4; ++m) {
        const __bf16* ap = Alds + (m * 16 + ln) * STRIDE_A + k0;
        v8bf alo = *(const v8bf*)ap;             // K = k0..k0+7
        v8bf ahi = *(const v8bf*)(ap + 8);       // K = k0+8..k0+15
        acc[m] = __builtin_amdgcn_wmma_f32_16x16x32_bf16(
            false, join16(alo, ahi), false, bfrag, (short)0, acc[m], false, false);
      }
    }

    // ---- phase A epilogue: z | r*h | cand-x per column region ----
    {
      const int col = n0 + ln;
      if (col < 512) {                           // z = sigmoid(g + bg)
        const float bgv = bg[col];
        #pragma unroll
        for (int m = 0; m < 4; ++m)
          #pragma unroll
          for (int i = 0; i < 8; ++i) {
            int row = m * 16 + i + lh * 8;
            zbuf[row * 512 + col] = fast_sigmoid(acc[m][i] + bgv);
          }
      } else if (col < 1024) {                   // rh = sigmoid(g + bg) * h
        const float bgv = bg[col];
        const int c2 = col - 512;
        #pragma unroll
        for (int m = 0; m < 4; ++m)
          #pragma unroll
          for (int i = 0; i < 8; ++i) {
            int row = m * 16 + i + lh * 8;
            float r = fast_sigmoid(acc[m][i] + bgv);
            rhbuf[row * 512 + c2] = (__bf16)(r * hbuf[row * 512 + c2]);
          }
      } else {                                   // cpre = x@Wc_x + bc
        const int c2 = col - 1024;
        const float bcv = bc[c2];
        #pragma unroll
        for (int m = 0; m < 4; ++m)
          #pragma unroll
          for (int i = 0; i < 8; ++i) {
            int row = m * 16 + i + lh * 8;
            cbuf[row * 512 + c2] = acc[m][i] + bcv;
          }
      }
    }

    target += NWG;
    grid_barrier(cnt, target);

    // ---- phase B: cand = tanh(cpre + rh @ Wc_h); h update (strips 0..31) ----
    if (strip < 32) {
      v8f acc2[4];
      #pragma unroll
      for (int m = 0; m < 4; ++m) acc2[m] = vzero;

      const __bf16* B2 = Wt2 + (size_t)(n0 + ln) * HID + lh * 16;
      #pragma unroll 4
      for (int kt = 0; kt < 16; ++kt) {          // K = 512
        v16bf bfrag = *(const v16bf*)(B2 + kt * 32);
        const int k0 = kt * 32 + lh * 16;
        #pragma unroll
        for (int m = 0; m < 4; ++m) {
          const __bf16* ap = rhbuf + (m * 16 + ln) * HID + k0;
          v8bf alo = *(const v8bf*)ap;
          v8bf ahi = *(const v8bf*)(ap + 8);
          acc2[m] = __builtin_amdgcn_wmma_f32_16x16x32_bf16(
              false, join16(alo, ahi), false, bfrag, (short)0, acc2[m], false, false);
        }
      }

      const int colb = n0 + ln;
      float* outT = out + (size_t)t * (BATCH * HID);
      #pragma unroll
      for (int m = 0; m < 4; ++m)
        #pragma unroll
        for (int i = 0; i < 8; ++i) {
          int row = m * 16 + i + lh * 8;
          int idx = row * 512 + colb;
          float cand = fast_tanh(acc2[m][i] + cbuf[idx]);
          float zz = zbuf[idx];
          float hp = hbuf[idx];
          float hn = (1.0f - zz) * hp + zz * cand;
          outT[idx] = hn;
          hbuf[idx] = hn;
          if (t == SEQ - 1)
            out[(size_t)SEQ * BATCH * HID + idx] = hn;   // h_last
        }
    }

    target += NWG;
    grid_barrier(cnt, target);
  }
}

// ---------------- launch ----------------
extern "C" void kernel_launch(void* const* d_in, const int* in_sizes, int n_in,
                              void* d_out, int out_size, void* d_ws, size_t ws_size,
                              hipStream_t stream) {
  const float* x  = (const float*)d_in[0];
  const float* h0 = (const float*)d_in[1];
  const float* Wg = (const float*)d_in[2];
  const float* bg = (const float*)d_in[3];
  const float* Wc = (const float*)d_in[4];
  const float* bc = (const float*)d_in[5];
  float* out = (float*)d_out;

  char* ws = (char*)d_ws;
  __bf16*   Wt1   = (__bf16*)(ws + WS_WT1);
  __bf16*   Wt2   = (__bf16*)(ws + WS_WT2);
  float*    hbuf  = (float*)(ws + WS_H);
  float*    zbuf  = (float*)(ws + WS_Z);
  float*    cbuf  = (float*)(ws + WS_C);
  __bf16*   rhbuf = (__bf16*)(ws + WS_RH);
  unsigned* cnt   = (unsigned*)(ws + WS_CNT);

  const int prepTotal = NA * KA + HID * HID + BATCH * HID + 1;
  gru_prep<<<(prepTotal + TPB - 1) / TPB, TPB, 0, stream>>>(Wg, Wc, h0, Wt1, Wt2, hbuf, cnt);

  const size_t ldsBytes = (size_t)64 * STRIDE_A * sizeof(__bf16);  // 132,096 B (<320 KB WGP LDS)
  (void)hipFuncSetAttribute(reinterpret_cast<const void*>(gru_scan),
                            hipFuncAttributeMaxDynamicSharedMemorySize, (int)ldsBytes);
  gru_scan<<<NWG, TPB, ldsBytes, stream>>>(x, bg, bc, Wt1, Wt2,
                                           hbuf, zbuf, cbuf, rhbuf, cnt, out);
}
